// MultiHeadAttention_54133767798978
// MI455X (gfx1250) — matrix operand order repaired and checked by reading, not compiled, not run on hardware
//
#include <hip/hip_runtime.h>
#include <hip/hip_bf16.h>

// ---------------------------------------------------------------------------
// MI455X (gfx1250) multi-head attention, bf16 WMMA (v_wmma_f32_16x16x32_bf16)
// DPP ROW_XMASK softmax reductions, packed b32 LDS transpose staging.
// ---------------------------------------------------------------------------

typedef __attribute__((ext_vector_type(16))) __bf16 v16bf;
typedef __attribute__((ext_vector_type(8)))  float  v8f;

union FragAB { v16bf v; unsigned int u[8]; };

__device__ __forceinline__ unsigned short f2bf(float f) {
    unsigned u = __float_as_uint(f);
    u += 0x7FFFu + ((u >> 16) & 1u);      // round-to-nearest-even
    return (unsigned short)(u >> 16);
}

// A-fragment K index of pair r (VGPR r holds K=kpair, kpair+1) for half-wave g.
__device__ __forceinline__ int kpair(int r, int g) {
    return g * 16 + 2 * r;
}

__device__ __forceinline__ v8f wmma_bf16(const FragAB& a, const FragAB& b, v8f c) {
    return __builtin_amdgcn_wmma_f32_16x16x32_bf16(false, a.v, false, b.v,
                                                   (short)0, c, false, false);
}

// DPP16 ROW_XMASK lane-xor shuffle (wave32): dpp_ctrl = 0x160 | xor_mask.
template <int CTRL>
__device__ __forceinline__ float dpp_xor(float x) {
    int r = __builtin_amdgcn_update_dpp(0, __float_as_int(x), CTRL, 0xf, 0xf, true);
    return __int_as_float(r);
}
__device__ __forceinline__ float xmax16(float x) {
    x = fmaxf(x, dpp_xor<0x161>(x));
    x = fmaxf(x, dpp_xor<0x162>(x));
    x = fmaxf(x, dpp_xor<0x164>(x));
    x = fmaxf(x, dpp_xor<0x168>(x));
    return x;
}
__device__ __forceinline__ float xsum16(float x) {
    x += dpp_xor<0x161>(x);
    x += dpp_xor<0x162>(x);
    x += dpp_xor<0x164>(x);
    x += dpp_xor<0x168>(x);
    return x;
}

// ---------------------------------------------------------------------------
// f32 -> bf16 conversion
// ---------------------------------------------------------------------------
__global__ void cvt_f32_bf16(const float* __restrict__ in,
                             unsigned short* __restrict__ out, int n) {
    int i = blockIdx.x * blockDim.x + threadIdx.x;
    if (i < n) out[i] = f2bf(in[i]);
}

// ---------------------------------------------------------------------------
// bf16 GEMM: C[M,N] = A[M,K] * B[K,N] (+bias). Block tile 128x128, BK=32.
// 8 waves: wave(wm,wn) computes a 64x32 sub-tile as 4x2 WMMA 16x16 tiles.
// MODE 0: scatter bf16 results into Q(scaled)/K/V in (B,H,N,hd) layout.
// MODE 1: f32 + bias to out.
// ---------------------------------------------------------------------------
template <int MODE>
__global__ __launch_bounds__(256)
void gemm_bf16(const unsigned short* __restrict__ A,
               const unsigned short* __restrict__ Bm,
               const float* __restrict__ bias,
               unsigned short* __restrict__ Qd,
               unsigned short* __restrict__ Kd,
               unsigned short* __restrict__ Vd,
               float* __restrict__ outF,
               int M, int N, int K, float qscale) {
    __shared__ __align__(16) unsigned short Alds[128 * 32];  // [row][k]
    __shared__ __align__(16) unsigned short Blds[128 * 32];  // transposed: [col][k]

    const int t    = threadIdx.x;
    const int lane = t & 31;
    const int w    = t >> 5;
    const int g    = lane >> 4;
    const int ln   = lane & 15;
    const int wm   = w >> 2;          // 0..1
    const int wn   = w & 3;           // 0..3
    const int mBase = blockIdx.y * 128;
    const int nBase = blockIdx.x * 128;

    v8f acc[4][2];
#pragma unroll
    for (int i = 0; i < 4; i++)
#pragma unroll
        for (int j = 0; j < 2; j++) acc[i][j] = {};

    for (int kk = 0; kk < K; kk += 32) {
        // ---- stage A tile (128x32) row-major: 512 uint4 chunks ----
#pragma unroll
        for (int i = 0; i < 2; i++) {
            int c = t + i * 256;
            int row = c >> 2, seg = c & 3;
            uint4 d = *(const uint4*)(A + (size_t)(mBase + row) * K + kk + seg * 8);
            *(uint4*)(&Alds[row * 32 + seg * 8]) = d;
        }
        // ---- stage B tile (32x128) transposed [col][k], packed b32 stores ----
        {
            int kr = (t >> 4) * 2;   // even k-row: 0,2,...,30
            int cs = t & 15;         // 8-column segment
            uint4 d0 = *(const uint4*)(Bm + (size_t)(kk + kr) * N + nBase + cs * 8);
            uint4 d1 = *(const uint4*)(Bm + (size_t)(kk + kr + 1) * N + nBase + cs * 8);
            const unsigned short* p0 = (const unsigned short*)&d0;
            const unsigned short* p1 = (const unsigned short*)&d1;
#pragma unroll
            for (int j = 0; j < 8; j++) {
                unsigned pk = (unsigned)p0[j] | ((unsigned)p1[j] << 16);
                *(unsigned*)(&Blds[(cs * 8 + j) * 32 + kr]) = pk;
            }
        }
        // prefetch next A slice into L2 while we compute (global_prefetch_b8)
        if (kk + 32 < K)
            __builtin_prefetch(A + (size_t)(mBase + (t >> 1)) * K + kk + 32, 0, 1);
        __syncthreads();

        FragAB af[4], bf[2];
#pragma unroll
        for (int rt = 0; rt < 4; rt++) {
            int rowb = wm * 64 + rt * 16 + ln;
#pragma unroll
            for (int r = 0; r < 8; r++)
                af[rt].u[r] = *(const unsigned int*)(&Alds[rowb * 32 + kpair(r, g)]);
        }
#pragma unroll
        for (int ct = 0; ct < 2; ct++) {
            int colb = wn * 32 + ct * 16 + ln;
#pragma unroll
            for (int r = 0; r < 8; r++)
                bf[ct].u[r] = *(const unsigned int*)(&Blds[colb * 32 + g * 16 + 2 * r]);
        }
#pragma unroll
        for (int rt = 0; rt < 4; rt++)
#pragma unroll
            for (int ct = 0; ct < 2; ct++)
                acc[rt][ct] = wmma_bf16(af[rt], bf[ct], acc[rt][ct]);
        __syncthreads();
    }

    // ---- epilogue ----
#pragma unroll
    for (int rt = 0; rt < 4; rt++) {
#pragma unroll
        for (int ct = 0; ct < 2; ct++) {
            int col = nBase + wn * 32 + ct * 16 + ln;
            float bv = bias[col];
#pragma unroll
            for (int r = 0; r < 8; r++) {
                int row  = mBase + wm * 64 + rt * 16 + g * 8 + r;
                float val = acc[rt][ct][r] + bv;
                if (MODE == 0) {
                    // col in [0,3D): tsel*D + h*64 + d ; row = b*N + n
                    int tsel = col >> 10;
                    int hh   = (col >> 6) & 15;
                    int d    = col & 63;
                    int b    = row >> 10;
                    int n    = row & 1023;
                    size_t dst = ((size_t)(b * 16 + hh) * 1024 + n) * 64 + d;
                    if (tsel == 0)      Qd[dst] = f2bf(val * qscale);
                    else if (tsel == 1) Kd[dst] = f2bf(val);
                    else                Vd[dst] = f2bf(val);
                } else {
                    outF[(size_t)row * N + col] = val;
                }
            }
        }
    }
}

// ---------------------------------------------------------------------------
// Flash attention: grid (N/128, B*H), 256 threads (8 waves).
// Wave w handles query rows [blockIdx.x*128 + w*16, +16). Key blocks of 32.
// Q pre-scaled by hd^-0.5. Output -> Ob in (B,N,H*hd) bf16 layout.
// ---------------------------------------------------------------------------
__global__ __launch_bounds__(256)
void attn_kernel(const unsigned short* __restrict__ Qd,
                 const unsigned short* __restrict__ Kd,
                 const unsigned short* __restrict__ Vd,
                 unsigned short* __restrict__ Ob) {
    __shared__ __align__(16) unsigned short Klds[32 * 64];      // [key][hd]
    __shared__ __align__(16) unsigned short Vlds[64 * 32];      // transposed [hd][key]
    __shared__ __align__(16) unsigned short Plds[8 * 16 * 32];  // per-wave P tile

    const int t    = threadIdx.x;
    const int lane = t & 31;
    const int w    = t >> 5;
    const int g    = lane >> 4;
    const int ln   = lane & 15;
    const int bh   = blockIdx.y;           // b*16 + h
    const int b    = bh >> 4;
    const int h    = bh & 15;
    const size_t base = (size_t)bh * 1024 * 64;
    const int q0 = blockIdx.x * 128 + w * 16;

    // Q fragments: 16 rows x 64 (hd) -> two 16x32 A-frags
    FragAB aq[2];
#pragma unroll
    for (int f = 0; f < 2; f++)
#pragma unroll
        for (int r = 0; r < 8; r++)
            aq[f].u[r] = *(const unsigned int*)(Qd + base + (size_t)(q0 + ln) * 64 +
                                                f * 32 + kpair(r, g));

    float m_r[8], l_r[8];
#pragma unroll
    for (int r = 0; r < 8; r++) { m_r[r] = -1e30f; l_r[r] = 0.0f; }
    v8f acc[4];
#pragma unroll
    for (int i = 0; i < 4; i++) acc[i] = {};

    for (int kb = 0; kb < 1024; kb += 32) {
        // ---- stage K block (32x64) row-major: one uint4 per thread ----
        {
            int krow = t >> 3, kseg = (t & 7) * 8;
            *(uint4*)(&Klds[krow * 64 + kseg]) =
                *(const uint4*)(Kd + base + (size_t)(kb + krow) * 64 + kseg);
        }
        // ---- stage V block transposed -> [hd][key], packed b32 stores ----
        if (t < 128) {
            int vk = (t >> 3) * 2;   // even key: 0,2,...,30
            int vh = (t & 7) * 8;    // 8-wide hd segment
            uint4 d0 = *(const uint4*)(Vd + base + (size_t)(kb + vk) * 64 + vh);
            uint4 d1 = *(const uint4*)(Vd + base + (size_t)(kb + vk + 1) * 64 + vh);
            const unsigned short* p0 = (const unsigned short*)&d0;
            const unsigned short* p1 = (const unsigned short*)&d1;
#pragma unroll
            for (int j = 0; j < 8; j++) {
                unsigned pk = (unsigned)p0[j] | ((unsigned)p1[j] << 16);
                *(unsigned*)(&Vlds[(vh + j) * 32 + vk]) = pk;
            }
        }
        __syncthreads();

        // ---- S = Q * K^T (16x32 per wave): 4 WMMAs ----
        v8f s0 = {}, s1 = {};
#pragma unroll
        for (int f = 0; f < 2; f++) {
            FragAB b0, b1;
#pragma unroll
            for (int r = 0; r < 8; r++) {
                b0.u[r] = *(const unsigned int*)(&Klds[ln * 64        + f * 32 + g * 16 + 2 * r]);
                b1.u[r] = *(const unsigned int*)(&Klds[(16 + ln) * 64 + f * 32 + g * 16 + 2 * r]);
            }
            s0 = wmma_bf16(aq[f], b0, s0);
            s1 = wmma_bf16(aq[f], b1, s1);
        }

        // ---- online softmax: DPP ROW_XMASK 16-lane reductions (pure VALU) ----
        float al[8];
#pragma unroll
        for (int r = 0; r < 8; r++) {
            float mx = xmax16(fmaxf(s0[r], s1[r]));
            float newm = fmaxf(m_r[r], mx);
            al[r] = __expf(m_r[r] - newm);
            float p0 = __expf(s0[r] - newm);
            float p1 = __expf(s1[r] - newm);
            s0[r] = p0; s1[r] = p1;
            float rs = xsum16(p0 + p1);
            l_r[r] = l_r[r] * al[r] + rs;
            m_r[r] = newm;
        }
#pragma unroll
        for (int t4 = 0; t4 < 4; t4++)
#pragma unroll
            for (int r = 0; r < 8; r++) acc[t4][r] *= al[r];

        // ---- re-fragment P: C-layout -> A-layout via per-wave LDS tile ----
        unsigned short* pw = &Plds[w * 16 * 32];
#pragma unroll
        for (int r = 0; r < 8; r++) {
            int row = g * 8 + r;
            pw[row * 32 + ln]      = f2bf(s0[r]);
            pw[row * 32 + 16 + ln] = f2bf(s1[r]);
        }
        FragAB ap;
#pragma unroll
        for (int r = 0; r < 8; r++)
            ap.u[r] = *(const unsigned int*)(&pw[ln * 32 + kpair(r, g)]);

        // ---- O += P * V : 4 WMMAs over hd column groups ----
#pragma unroll
        for (int t4 = 0; t4 < 4; t4++) {
            FragAB bv;
#pragma unroll
            for (int r = 0; r < 8; r++)
                bv.u[r] = *(const unsigned int*)(&Vlds[(t4 * 16 + ln) * 32 + g * 16 + 2 * r]);
            acc[t4] = wmma_bf16(ap, bv, acc[t4]);
        }
        __syncthreads();
    }

    // ---- normalize and store bf16 into (B, N, H*hd) ----
#pragma unroll
    for (int r = 0; r < 8; r++) l_r[r] = 1.0f / l_r[r];
#pragma unroll
    for (int t4 = 0; t4 < 4; t4++)
#pragma unroll
        for (int r = 0; r < 8; r++) {
            int row = q0 + g * 8 + r;
            int col = h * 64 + t4 * 16 + ln;
            Ob[((size_t)b * 1024 + row) * 1024 + col] = f2bf(acc[t4][r] * l_r[r]);
        }
}

// ---------------------------------------------------------------------------
// launch
// ---------------------------------------------------------------------------
extern "C" void kernel_launch(void* const* d_in, const int* in_sizes, int n_in,
                              void* d_out, int out_size, void* d_ws, size_t ws_size,
                              hipStream_t stream) {
    const float* x      = (const float*)d_in[0];  // (8,1024,1024)
    const float* W_qkv  = (const float*)d_in[1];  // (1024,3072)
    const float* b_qkv  = (const float*)d_in[2];  // (3072,)
    const float* W_proj = (const float*)d_in[3];  // (1024,1024)
    const float* b_proj = (const float*)d_in[4];  // (1024,)
    float* out = (float*)d_out;                   // (8,1024,1024) f32

    const int NX  = 8 * 1024 * 1024;
    const int NWQ = 1024 * 3072;
    const int NWP = 1024 * 1024;
    const int NQKV = 8 * 16 * 1024 * 64;  // per Q/K/V tensor

    unsigned short* Xb    = (unsigned short*)d_ws;
    unsigned short* Wqkvb = Xb + NX;
    unsigned short* Wprjb = Wqkvb + NWQ;
    unsigned short* Qs    = Wprjb + NWP;
    unsigned short* Kb    = Qs + NQKV;
    unsigned short* Vb    = Kb + NQKV;
    unsigned short* Obf   = Vb + NQKV;

    const float SCALE = 0.125f;  // 64^-0.5

    cvt_f32_bf16<<<(NX + 255) / 256, 256, 0, stream>>>(x, Xb, NX);
    cvt_f32_bf16<<<(NWQ + 255) / 256, 256, 0, stream>>>(W_qkv, Wqkvb, NWQ);
    cvt_f32_bf16<<<(NWP + 255) / 256, 256, 0, stream>>>(W_proj, Wprjb, NWP);

    // QKV projection: (8192x1024)x(1024x3072), scatter into Q(scaled)/K/V
    gemm_bf16<0><<<dim3(24, 64), 256, 0, stream>>>(
        Xb, Wqkvb, b_qkv, Qs, Kb, Vb, nullptr, 8192, 3072, 1024, SCALE);

    // attention: grid (qchunks=8, B*H=128)
    attn_kernel<<<dim3(8, 128), 256, 0, stream>>>(Qs, Kb, Vb, Obf);

    // output projection: (8192x1024)x(1024x1024) + b_proj -> f32 out
    gemm_bf16<1><<<dim3(8, 64), 256, 0, stream>>>(
        Obf, Wprjb, b_proj, nullptr, nullptr, nullptr, out, 8192, 1024, 1024, 1.0f);
}
